// RelativeScaledDotProductAttention_46205258170723
// MI455X (gfx1250) — matrix operand order repaired and checked by reading, not compiled, not run on hardware
//
#include <hip/hip_runtime.h>

// Relative scaled-dot-product attention, fused flash-style kernel for gfx1250.
//
// Tile: workgroup = 16 waves (512 thr) = [16 batches] x [16 queries] x [k chunks of 32].
//   wave w: content GEMMs for batch b0+w (M=q),  relative GEMMs for query q0+w (M=b).
// All 4 contractions are v_wmma_f32_16x16x32_f16 (f32 accumulate).
// attn is written unnormalized (exp(s/8)); row sums go to d_ws; kernel 2 rescales.
//
// Memory plan: q/k/v (48 MB) stay hot in the 192 MB L2 (b-tile is the inner grid
// dim, so the 4 workgroups sharing one a_key/a_value q-slice are co-resident);
// a_key/a_value (512 MB) stream from HBM exactly once; attn (268 MB) written once
// here + one read/write rescale pass.  ~1.4 GB @ 23.3 TB/s ~= 60 us, memory-bound.

#define BATCH 64
#define SEQ   1024
#define DIM   64
#define TB    16
#define TQ    16
#define KC    32
#define NCHUNK (SEQ / KC)

typedef __attribute__((ext_vector_type(16))) _Float16 v16h;
typedef __attribute__((ext_vector_type(4)))  _Float16 v4h;
typedef __attribute__((ext_vector_type(8)))  float    v8f;

static __device__ __forceinline__ v8f wmma_f16(v16h a, v16h b, v8f c) {
  // D = A(16x32) * B(32x16) + C, f32 accumulate
  return __builtin_amdgcn_wmma_f32_16x16x32_f16(false, a, false, b, (short)0, c, false, false);
}

// K-index inside a 16x32 f16 A-fragment for element j (0..15) of lane-group grp (0/1).
// Contiguous order: lanes 0-15: K=0..15 ; lanes 16-31: K=16..31 (matches B fills).
static __device__ __forceinline__ int a_kidx(int grp, int j) {
  return 16 * grp + j;
}

// Build a 32(K)x16(N) B-fragment from 16 d-contiguous floats per lane (b128 loads).
static __device__ __forceinline__ v16h load_b_frag_contig(const float* __restrict__ p) {
  v16h b;
#pragma unroll
  for (int j4 = 0; j4 < 4; ++j4) {
    const float4 t = *(const float4*)(p + 4 * j4);
    b[4 * j4 + 0] = (_Float16)t.x;
    b[4 * j4 + 1] = (_Float16)t.y;
    b[4 * j4 + 2] = (_Float16)t.z;
    b[4 * j4 + 3] = (_Float16)t.w;
  }
  return b;
}

__global__ __launch_bounds__(512)
void relattn_fused(const float* __restrict__ Q, const float* __restrict__ K,
                   const float* __restrict__ V, const float* __restrict__ AK,
                   const float* __restrict__ AV,
                   float* __restrict__ out, float* __restrict__ attn,
                   float* __restrict__ row_l) {
  __shared__ float    S_lds[TB * TQ * KC];   // 32 KB: score block [b][q][k]
  __shared__ _Float16 P_lds[TB * TQ * KC];   // 16 KB: exp(scores) as f16, [b][q][k]
  __shared__ float    l_lds[TB * TQ];        // 1 KB : row sums

  const int lane = threadIdx.x & 31;
  const int w    = threadIdx.x >> 5;   // wave id 0..15
  const int grp  = lane >> 4;          // lane group (0/1)
  const int ln   = lane & 15;

  const int b0  = blockIdx.x * TB;     // 4 b-tiles (inner -> a_key slice shared in L2)
  const int q0  = blockIdx.y * TQ;     // 64 q-tiles
  const int myb = b0 + w;
  const int myq = q0 + w;

  // ---- persistent A operands (query slices), two d-halves each ----
  v16h Acq[2];  // content A: rows = queries q0..q0+15 of batch myb
  v16h Arq[2];  // relative A: rows = batches b0..b0+15 at query myq
#pragma unroll
  for (int h = 0; h < 2; ++h) {
    const float* qc = &Q[((size_t)myb * SEQ + (q0 + ln)) * DIM + 32 * h];
    const float* qr = &Q[((size_t)(b0 + ln) * SEQ + myq) * DIM + 32 * h];
#pragma unroll
    for (int j = 0; j < 16; ++j) {
      const int d = a_kidx(grp, j);
      Acq[h][j] = (_Float16)qc[d];
      Arq[h][j] = (_Float16)qr[d];
    }
  }

  v8f acc_c[4], acc_r[4];   // output accumulators, 4 d-tiles of 16
#pragma unroll
  for (int i = 0; i < 4; ++i) {
    acc_c[i] = (v8f){0.f, 0.f, 0.f, 0.f, 0.f, 0.f, 0.f, 0.f};
    acc_r[i] = (v8f){0.f, 0.f, 0.f, 0.f, 0.f, 0.f, 0.f, 0.f};
  }
  float psum[4] = {0.f, 0.f, 0.f, 0.f};   // per-lane partial row sums (exp phase)

  // exp-phase mapping: lane covers row q = 4*pass + lane/8, cols kk = (lane%8)*4 ..+3
  const int eq_g  = lane >> 3;        // 0..3
  const int eq_kk = (lane & 7) * 4;   // 0,4,..,28

  __syncthreads();

  for (int c = 0; c < NCHUNK; ++c) {
    const int k0 = c * KC;

    // prefetch next chunk's streaming operands (global_prefetch_b8)
    if (c + 1 < NCHUNK) {
      const int kn = k0 + KC;
      __builtin_prefetch(&K [((size_t)myb * SEQ + kn + lane) * DIM], 0, 0);
      __builtin_prefetch(&AK[((size_t)myq * SEQ + kn + lane) * DIM], 0, 0);
      __builtin_prefetch(&V [((size_t)myb * SEQ + kn + lane) * DIM], 0, 0);
      __builtin_prefetch(&AV[((size_t)myq * SEQ + kn + lane) * DIM], 0, 0);
    }

    // ---- (1) content scores: qk[q, k-tile] for batch myb ----
#pragma unroll
    for (int kt = 0; kt < 2; ++kt) {
      v8f sc = (v8f){0.f, 0.f, 0.f, 0.f, 0.f, 0.f, 0.f, 0.f};
#pragma unroll
      for (int h = 0; h < 2; ++h) {
        const v16h Bk = load_b_frag_contig(
            &K[((size_t)myb * SEQ + (k0 + kt * 16 + ln)) * DIM + 32 * h + 16 * grp]);
        sc = wmma_f16(Acq[h], Bk, sc);
      }
#pragma unroll
      for (int r = 0; r < 8; ++r)
        S_lds[(w * TQ + (r + 8 * grp)) * KC + kt * 16 + ln] = sc[r];
    }
    __syncthreads();

    // ---- (2) relative scores: rel[b, k-tile] for query myq, added in ----
#pragma unroll
    for (int kt = 0; kt < 2; ++kt) {
      v8f sr = (v8f){0.f, 0.f, 0.f, 0.f, 0.f, 0.f, 0.f, 0.f};
#pragma unroll
      for (int h = 0; h < 2; ++h) {
        const v16h Ba = load_b_frag_contig(
            &AK[((size_t)myq * SEQ + (k0 + kt * 16 + ln)) * DIM + 32 * h + 16 * grp]);
        sr = wmma_f16(Arq[h], Ba, sr);
      }
#pragma unroll
      for (int r = 0; r < 8; ++r)
        S_lds[((r + 8 * grp) * TQ + w) * KC + kt * 16 + ln] += sr[r];
    }
    __syncthreads();

    // ---- (3) exp + attn store (b128, coalesced) + f16 P; row sums stay in regs ----
    // scores ~ N(0,2) after the 1/8 scale -> exp without max-subtraction is safe in f32
#pragma unroll
    for (int p = 0; p < 4; ++p) {
      const int q = 4 * p + eq_g;
      const float4 s = *(const float4*)&S_lds[(w * TQ + q) * KC + eq_kk];
      const float e0 = __expf(s.x * 0.125f);
      const float e1 = __expf(s.y * 0.125f);
      const float e2 = __expf(s.z * 0.125f);
      const float e3 = __expf(s.w * 0.125f);
      psum[p] += (e0 + e1) + (e2 + e3);
      *(float4*)&attn[((size_t)myb * SEQ + (q0 + q)) * SEQ + k0 + eq_kk] =
          make_float4(e0, e1, e2, e3);
      v4h ph;
      ph[0] = (_Float16)e0; ph[1] = (_Float16)e1;
      ph[2] = (_Float16)e2; ph[3] = (_Float16)e3;
      *(v4h*)&P_lds[(w * TQ + q) * KC + eq_kk] = ph;
    }
    __syncthreads();

    // ---- (4) output GEMMs: acc_c += P[b] @ V[b],  acc_r += P[:,q] @ AV[q] ----
    v16h Pa_c, Pa_r;
#pragma unroll
    for (int j = 0; j < 16; ++j) {
      const int kk = a_kidx(grp, j);
      Pa_c[j] = P_lds[(w * TQ + ln) * KC + kk];   // rows = queries (batch myb)
      Pa_r[j] = P_lds[(ln * TQ + w) * KC + kk];   // rows = batches (query myq)
    }
#pragma unroll
    for (int dt = 0; dt < 4; ++dt) {
      v16h Bv, Bav;
#pragma unroll
      for (int j = 0; j < 16; ++j) {
        const int kk = 16 * grp + j;
        Bv[j]  = (_Float16)V [((size_t)myb * SEQ + k0 + kk) * DIM + dt * 16 + ln];
        Bav[j] = (_Float16)AV[((size_t)myq * SEQ + k0 + kk) * DIM + dt * 16 + ln];
      }
      acc_c[dt] = wmma_f16(Pa_c, Bv,  acc_c[dt]);
      acc_r[dt] = wmma_f16(Pa_r, Bav, acc_r[dt]);
    }
    // no barrier needed here: next chunk's P_lds write is fenced by barriers (1)+(2)
  }

  // ---- final row-sum reduction: 8 lanes (same eq_g) share a row ----
#pragma unroll
  for (int p = 0; p < 4; ++p) {
    float r = psum[p];
    r += __shfl_xor(r, 1);
    r += __shfl_xor(r, 2);
    r += __shfl_xor(r, 4);
    if ((lane & 7) == 0) l_lds[w * TQ + 4 * p + eq_g] = r;
  }
  __syncthreads();

  // ---- epilogue: combine content + relative outputs via LDS (reuse S_lds), ----
  // ---- normalize by row sums, write out. Two 32-wide d-halves fit in 32 KB. ----
#pragma unroll
  for (int half = 0; half < 2; ++half) {
    // relative wave (owns query w) deposits its half: S_lds[b][q=w][d(32)]
#pragma unroll
    for (int dt = 2 * half; dt < 2 * half + 2; ++dt)
#pragma unroll
      for (int r = 0; r < 8; ++r) {
        const int bb = r + 8 * grp;
        S_lds[(bb * TQ + w) * 32 + (dt & 1) * 16 + ln] = acc_r[dt][r];
      }
    __syncthreads();
    // content wave (owns batch w) adds its half, scales by 1/rowsum, stores
#pragma unroll
    for (int dt = 2 * half; dt < 2 * half + 2; ++dt)
#pragma unroll
      for (int r = 0; r < 8; ++r) {
        const int qq  = r + 8 * grp;
        const float l = l_lds[w * TQ + qq];
        const float v = (acc_c[dt][r] + S_lds[(w * TQ + qq) * 32 + (dt & 1) * 16 + ln]) * (1.0f / l);
        out[((size_t)myb * SEQ + q0 + qq) * DIM + dt * 16 + ln] = v;
      }
    __syncthreads();
  }

  // stash row sums for the attn rescale pass
  if (threadIdx.x < TB * TQ) {
    const int bb = threadIdx.x / TQ, qq = threadIdx.x % TQ;
    row_l[(size_t)(b0 + bb) * SEQ + (q0 + qq)] = l_lds[threadIdx.x];
  }
}

// Pass 2: attn[b,q,k] *= 1/rowsum[b,q]  (float4, row-aligned since SEQ % 4 == 0)
__global__ __launch_bounds__(256)
void attn_rescale(float* __restrict__ attn, const float* __restrict__ row_l,
                  size_t n4) {
  size_t i = (size_t)blockIdx.x * blockDim.x + threadIdx.x;
  const size_t stride = (size_t)gridDim.x * blockDim.x;
  float4* a4 = (float4*)attn;
  for (; i < n4; i += stride) {
    const size_t row = i / (SEQ / 4);
    const float inv = 1.0f / row_l[row];
    float4 v = a4[i];
    v.x *= inv; v.y *= inv; v.z *= inv; v.w *= inv;
    a4[i] = v;
  }
}

extern "C" void kernel_launch(void* const* d_in, const int* in_sizes, int n_in,
                              void* d_out, int out_size, void* d_ws, size_t ws_size,
                              hipStream_t stream) {
  const float* Q  = (const float*)d_in[0];
  const float* K  = (const float*)d_in[1];
  const float* V  = (const float*)d_in[2];
  const float* AK = (const float*)d_in[3];
  const float* AV = (const float*)d_in[4];

  float* out   = (float*)d_out;                       // [B,S,D]
  float* attn  = out + (size_t)BATCH * SEQ * DIM;     // [B,S,S]
  float* row_l = (float*)d_ws;                        // [B,S] row sums

  dim3 grid(BATCH / TB, SEQ / TQ);   // x = b-tile (inner: shares a_key slice in L2)
  relattn_fused<<<grid, 512, 0, stream>>>(Q, K, V, AK, AV, out, attn, row_l);

  const size_t n4 = (size_t)BATCH * SEQ * SEQ / 4;
  attn_rescale<<<4096, 256, 0, stream>>>(attn, row_l, n4);
}
